// FastMoEExperts_6640019440352
// MI455X (gfx1250) — matrix operand order repaired and checked by reading, not compiled, not run on hardware
//
#include <hip/hip_runtime.h>
#include <hip/hip_bf16.h>

typedef __attribute__((ext_vector_type(16))) __bf16 v16bf;
typedef __attribute__((ext_vector_type(8)))  __bf16 v8bf;
typedef __attribute__((ext_vector_type(4)))  __bf16 v4bf;
typedef __attribute__((ext_vector_type(8)))  float  v8f;

#define E_  8
#define D_  128
#define H_  512
#define MTOT (8 * 2048)          // B*T tokens
#define WELEM (E_ * H_ * D_)     // 524288 elements per weight tensor

// ---------------------------------------------------------------------------
// Kernel 1: deterministic per-block abs-sum partials for the 3 weight tensors
// ---------------------------------------------------------------------------
__global__ __launch_bounds__(256)
void absum_partial(const float* __restrict__ g, const float* __restrict__ u,
                   const float* __restrict__ d, float* __restrict__ partial) {
    const float* src = (blockIdx.y == 0) ? g : (blockIdx.y == 1) ? u : d;
    __shared__ float sm[256];
    int base = blockIdx.x * 2048 + threadIdx.x;
    float s = 0.f;
#pragma unroll
    for (int i = 0; i < 8; ++i) s += fabsf(src[base + i * 256]);
    sm[threadIdx.x] = s;
    __syncthreads();
    for (int off = 128; off > 0; off >>= 1) {
        if (threadIdx.x < (unsigned)off) sm[threadIdx.x] += sm[threadIdx.x + off];
        __syncthreads();
    }
    if (threadIdx.x == 0) partial[blockIdx.y * 256 + blockIdx.x] = sm[0];
}

// ---------------------------------------------------------------------------
// Kernel 2: serial (deterministic) finalize: gamma[t] = sum/N + 1e-8
// ---------------------------------------------------------------------------
__global__ void gamma_final(const float* __restrict__ partial, float* __restrict__ hdr) {
    int t = threadIdx.x;
    if (t < 3) {
        float s = 0.f;
        for (int i = 0; i < 256; ++i) s += partial[t * 256 + i];
        hdr[t] = s / (float)WELEM + 1e-8f;
    }
}

// ---------------------------------------------------------------------------
// Kernel 3: ternary-quantize weights to bf16 {-1, 0, +1} (scale applied later)
// ---------------------------------------------------------------------------
__global__ __launch_bounds__(256)
void quantize_tern(const float* __restrict__ g, const float* __restrict__ u,
                   const float* __restrict__ d, const float* __restrict__ hdr,
                   __bf16* __restrict__ q) {
    int t = blockIdx.y;
    const float* src = (t == 0) ? g : (t == 1) ? u : d;
    float gamma = hdr[t];
    int i = blockIdx.x * 256 + threadIdx.x;
    float v = __builtin_rintf(src[i] / gamma);       // round-half-even, like jnp.round
    v = fminf(1.f, fmaxf(-1.f, v));
    q[(size_t)t * WELEM + i] = (__bf16)v;
}

// ---------------------------------------------------------------------------
// Kernel 4: fused BitNet MoE expert kernel, M-blocked 2x per wave.
// grid (MTOT/128, E), block 128 (4 waves). Each wave owns 32 tokens
// (two 16-row WMMA M-subtiles) so every B fragment feeds two WMMAs.
// ---------------------------------------------------------------------------
__global__ __launch_bounds__(128)
void moe_fused(const float* __restrict__ x, const float* __restrict__ ew,
               const __bf16* __restrict__ gq, const __bf16* __restrict__ uq,
               const __bf16* __restrict__ dq, const float* __restrict__ hdr,
               float* __restrict__ out) {
    __shared__ __bf16 xs[128][136];     // 128 tokens x 128 D (+8 pad, conflict-free)
    __shared__ __bf16 hs[4][32][136];   // per-wave hidden chunk: 32 tokens x 128 H
    __shared__ float  ews[128];

    const int e    = blockIdx.y;
    const int m0w  = blockIdx.x * 128;
    const int tid  = threadIdx.x;
    const int wave = tid >> 5;
    const int lane = tid & 31;
    const int lrow = lane & 15;
    const int lhi  = lane >> 4;          // 0 or 1 (half-wave)
    const int mA   = wave * 32;          // subtile A row base within tile
    const int mB   = mA + 16;            // subtile B row base within tile

    // ---- cooperative: load x tile, convert to bf16, stage in LDS ----
#pragma unroll
    for (int i = 0; i < 32; ++i) {
        int f   = tid + i * 128;         // 4096 float4 total
        int row = f >> 5;                // 32 float4 per 128-float row
        int c4  = f & 31;
        const float4 v = *(const float4*)(x + ((size_t)(m0w + row) * E_ + e) * D_ + c4 * 4);
        v4bf b;
        b[0] = (__bf16)v.x; b[1] = (__bf16)v.y; b[2] = (__bf16)v.z; b[3] = (__bf16)v.w;
        *(v4bf*)&xs[row][c4 * 4] = b;
    }
    ews[tid] = ew[(size_t)(m0w + tid) * E_ + e];
    __syncthreads();

    const float gg = hdr[0], gu = hdr[1], gd = hdr[2];

    const v8f vzero = {0.f,0.f,0.f,0.f,0.f,0.f,0.f,0.f};
    v8f acc0[8], acc1[8];                // down-proj accumulators, held across chunks
#pragma unroll
    for (int i = 0; i < 8; ++i) { acc0[i] = vzero; acc1[i] = vzero; }

    for (int chunk = 0; chunk < 4; ++chunk) {
        const int hch = chunk * 128;

        // A fragments for stage 1 (K = D = 128), both M-subtiles.
        // 16-bit A 16x32: lanes 0-15 hold K 0..15; lanes 16-31 hold K 16..31 (contiguous, matches B)
        v16bf a1a[4], a1b[4];
#pragma unroll
        for (int kk = 0; kk < 4; ++kk) {
            const __bf16* pa = &xs[mA + lrow][kk * 32 + lhi * 16];
            const __bf16* pb = &xs[mB + lrow][kk * 32 + lhi * 16];
            v8bf lo = *(const v8bf*)pa;
            v8bf hi = *(const v8bf*)(pa + 8);
            a1a[kk] = __builtin_shufflevector(lo, hi, 0,1,2,3,4,5,6,7,8,9,10,11,12,13,14,15);
            lo = *(const v8bf*)pb;
            hi = *(const v8bf*)(pb + 8);
            a1b[kk] = __builtin_shufflevector(lo, hi, 0,1,2,3,4,5,6,7,8,9,10,11,12,13,14,15);
        }

        // -------- stage 1: gate & up for this H chunk --------
        for (int ct = 0; ct < 8; ++ct) {
            const int hb = hch + ct * 16;
            // 16-bit B 32x16 as used here: lane n = W row; K contiguous (k = 16*lhi + s, same as A), 32B per lane
            const __bf16* gb = gq + ((size_t)e * H_ + hb + lrow) * D_ + lhi * 16;
            const __bf16* ub = uq + ((size_t)e * H_ + hb + lrow) * D_ + lhi * 16;
            v8f g0 = vzero, g1 = vzero, u0 = vzero, u1 = vzero;
#pragma unroll
            for (int kk = 0; kk < 4; ++kk) {
                v16bf bg = *(const v16bf*)(gb + kk * 32);
                v16bf bu = *(const v16bf*)(ub + kk * 32);
                g0 = __builtin_amdgcn_wmma_f32_16x16x32_bf16(false, a1a[kk], false, bg,
                                                             (short)0, g0, false, false);
                g1 = __builtin_amdgcn_wmma_f32_16x16x32_bf16(false, a1b[kk], false, bg,
                                                             (short)0, g1, false, false);
                u0 = __builtin_amdgcn_wmma_f32_16x16x32_bf16(false, a1a[kk], false, bu,
                                                             (short)0, u0, false, false);
                u1 = __builtin_amdgcn_wmma_f32_16x16x32_bf16(false, a1b[kk], false, bu,
                                                             (short)0, u1, false, false);
            }
            // SwiGLU; C/D layout: VGPR j -> row j + 8*lhi, lane&15 -> column
#pragma unroll
            for (int j = 0; j < 8; ++j) {
                float gv = gg * g0[j];
                float uv = gu * u0[j];
                float hv = (gv / (1.f + __expf(-gv))) * uv;
                hs[wave][j + lhi * 8][ct * 16 + lrow] = (__bf16)hv;
                gv = gg * g1[j];
                uv = gu * u1[j];
                hv = (gv / (1.f + __expf(-gv))) * uv;
                hs[wave][16 + j + lhi * 8][ct * 16 + lrow] = (__bf16)hv;
            }
        }
        asm volatile("s_wait_dscnt 0" ::: "memory");   // hidden stores -> A-frag loads

        // -------- stage 2: down GEMM partial over this K chunk of 128 --------
        v16bf a2a[4], a2b[4];
#pragma unroll
        for (int kk = 0; kk < 4; ++kk) {
            const __bf16* pa = &hs[wave][lrow][kk * 32 + lhi * 16];
            const __bf16* pb = &hs[wave][16 + lrow][kk * 32 + lhi * 16];
            v8bf lo = *(const v8bf*)pa;
            v8bf hi = *(const v8bf*)(pa + 8);
            a2a[kk] = __builtin_shufflevector(lo, hi, 0,1,2,3,4,5,6,7,8,9,10,11,12,13,14,15);
            lo = *(const v8bf*)pb;
            hi = *(const v8bf*)(pb + 8);
            a2b[kk] = __builtin_shufflevector(lo, hi, 0,1,2,3,4,5,6,7,8,9,10,11,12,13,14,15);
        }
#pragma unroll
        for (int dt = 0; dt < 8; ++dt) {
            const __bf16* db = dq + ((size_t)e * D_ + dt * 16 + lrow) * H_ + hch + lhi * 16;
#pragma unroll
            for (int kk = 0; kk < 4; ++kk) {
                v16bf bd = *(const v16bf*)(db + kk * 32);
                acc0[dt] = __builtin_amdgcn_wmma_f32_16x16x32_bf16(false, a2a[kk], false, bd,
                                                                   (short)0, acc0[dt], false, false);
                acc1[dt] = __builtin_amdgcn_wmma_f32_16x16x32_bf16(false, a2b[kk], false, bd,
                                                                   (short)0, acc1[dt], false, false);
            }
        }
    }

    // ---- epilogue: scale by gamma_down and routing weight, write fp32 ----
#pragma unroll
    for (int dt = 0; dt < 8; ++dt) {
#pragma unroll
        for (int j = 0; j < 8; ++j) {
            int r  = j + lhi * 8;
            int c  = dt * 16 + lrow;
            int m0 = m0w + mA + r;
            int m1 = m0w + mB + r;
            out[((size_t)m0 * E_ + e) * D_ + c] = acc0[dt][j] * gd * ews[mA + r];
            out[((size_t)m1 * E_ + e) * D_ + c] = acc1[dt][j] * gd * ews[mB + r];
        }
    }
}

// ---------------------------------------------------------------------------
extern "C" void kernel_launch(void* const* d_in, const int* in_sizes, int n_in,
                              void* d_out, int out_size, void* d_ws, size_t ws_size,
                              hipStream_t stream) {
    (void)in_sizes; (void)n_in; (void)out_size; (void)ws_size;
    const float* x  = (const float*)d_in[0];   // [B,T,E,D]
    const float* ew = (const float*)d_in[1];   // [B,T,E]
    const float* gw = (const float*)d_in[2];   // [E,H,D]
    const float* uw = (const float*)d_in[3];   // [E,H,D]
    const float* dw = (const float*)d_in[4];   // [E,D,H]
    float* out = (float*)d_out;

    char*  ws      = (char*)d_ws;
    float* hdr     = (float*)ws;               // 3 gammas
    float* partial = (float*)(ws + 64);        // 768 partial sums
    __bf16* q      = (__bf16*)(ws + 4096);     // 3 * 524288 bf16 ternary weights (~3 MB)

    absum_partial<<<dim3(256, 3), 256, 0, stream>>>(gw, uw, dw, partial);
    gamma_final<<<1, 32, 0, stream>>>(partial, hdr);
    quantize_tern<<<dim3(2048, 3), 256, 0, stream>>>(gw, uw, dw, hdr, q);
    moe_fused<<<dim3(MTOT / 128, E_), 128, 0, stream>>>(x, ew, q, q + WELEM, q + 2 * WELEM,
                                                        hdr, out);
}
